// BaseLayer_27058293964888
// MI455X (gfx1250) — matrix operand order repaired and checked by reading, not compiled, not run on hardware
//
#include <hip/hip_runtime.h>

#define NR   16384
#define FEAT 256
#define EMB  128
#define ODIM 128

typedef __attribute__((ext_vector_type(16))) _Float16 v16h;
typedef __attribute__((ext_vector_type(8)))  _Float16 v8h;
typedef __attribute__((ext_vector_type(8)))  float    v8f;
typedef __attribute__((ext_vector_type(4)))  float    v4f;

#if __has_builtin(__builtin_amdgcn_sched_barrier)
#define SCHED_FENCE() __builtin_amdgcn_sched_barrier(0)
#else
#define SCHED_FENCE() asm volatile("" ::: "memory")
#endif

static __device__ __forceinline__ v8f zero8() {
    v8f z;
#pragma unroll
    for (int i = 0; i < 8; ++i) z[i] = 0.0f;
    return z;
}

static __device__ __forceinline__ v16h cat8(v8h lo, v8h hi) {
    v16h r;
#pragma unroll
    for (int i = 0; i < 8; ++i) { r[i] = lo[i]; r[i + 8] = hi[i]; }
    return r;
}

static __device__ __forceinline__ v8h cvt8(v4f a, v4f b) {
    v8h r;
#pragma unroll
    for (int i = 0; i < 4; ++i) { r[i] = (_Float16)a[i]; r[i + 4] = (_Float16)b[i]; }
    return r;
}

// A-fragment (16x32 f16): lane l -> row l16, K-run kb+16*hi..+16 contiguous
static __device__ __forceinline__ v16h load_a16(const _Float16* base, int stride,
                                                int l16, int lhalf, int kb) {
    const _Float16* p = base + (size_t)l16 * stride + kb + 16 * lhalf;
    v8h lo = *(const v8h*)(p);
    v8h hi = *(const v8h*)(p + 8);
    return cat8(lo, hi);
}

// Same A-fragment pattern, converting from fp32 source on the fly
static __device__ __forceinline__ v16h load_a32(const float* base, int stride,
                                                int l16, int lhalf, int kb) {
    const float* p = base + (size_t)l16 * stride + kb + 16 * lhalf;
    v4f a0 = *(const v4f*)(p);      v4f a1 = *(const v4f*)(p + 4);
    v4f b0 = *(const v4f*)(p + 8);  v4f b1 = *(const v4f*)(p + 12);
    return cat8(cvt8(a0, a1), cvt8(b0, b1));
}

// B-fragment (32x16 f16): lane l holds column l16, K-run 16*hi..+16 contiguous at p
static __device__ __forceinline__ v16h load_b(const _Float16* p) {
    v8h lo = *(const v8h*)(p);
    v8h hi = *(const v8h*)(p + 8);
    return cat8(lo, hi);
}

static __device__ __forceinline__ v8f wmma_f16(v16h a, v16h b, v8f c) {
    return __builtin_amdgcn_wmma_f32_16x16x32_f16(false, a, false, b, (short)0, c,
                                                  false, false);
}

// CDNA5 async global->LDS copy (16B per lane), tracked by ASYNCcnt
static __device__ __forceinline__ void async_ld16(void* lds_dst, const void* gsrc) {
    unsigned l = (unsigned)(uintptr_t)lds_dst;  // generic addr low 32 bits = LDS offset
    asm volatile("global_load_async_to_lds_b128 %0, %1, off"
                 :: "v"(l), "v"(gsrc) : "memory");
}
static __device__ __forceinline__ void wait_async() {
    asm volatile("s_wait_asynccnt 0x0" ::: "memory");
}

// ---------------- h = x @ W1 + b1 (emits h fp32, h f16, h^T f16) ----------------
__global__ void __launch_bounds__(256)
k_hidden(const float* __restrict__ x, const float* __restrict__ W1,
         const float* __restrict__ b1, float* __restrict__ h32,
         _Float16* __restrict__ h16, _Float16* __restrict__ h16t) {
    __shared__ __align__(16) _Float16 w1t[EMB][FEAT / 2 + 8];
    __shared__ float b1s[EMB];

    const int tid = threadIdx.x;
    const int lane = tid & 31, wave = tid >> 5;
    const int l16 = lane & 15, lhalf = lane >> 4;
    const int m0 = blockIdx.x * 128 + wave * 16;

    if (tid < EMB) b1s[tid] = b1[tid];

    v8f acc[8];
#pragma unroll
    for (int nc = 0; nc < 8; ++nc) acc[nc] = zero8();

    for (int half = 0; half < 2; ++half) {
        __syncthreads();
        for (int idx = tid; idx < 128 * EMB; idx += 256) {
            int n = idx & 127;
            int kk = idx >> 7;
            w1t[n][kk] = (_Float16)W1[(size_t)(half * 128 + kk) * EMB + n];
        }
        __syncthreads();
#pragma unroll
        for (int kc = 0; kc < 4; ++kc) {
            v16h a = load_a32(x + (size_t)m0 * FEAT, FEAT, l16, lhalf,
                              half * 128 + kc * 32);
            v16h bb[8];
#pragma unroll
            for (int nc = 0; nc < 8; ++nc)
                bb[nc] = load_b(&w1t[nc * 16 + l16][kc * 32 + 16 * lhalf]);
            SCHED_FENCE();  // keep all 8 B-frags live; WMMAs issue back-to-back
#pragma unroll
            for (int nc = 0; nc < 8; ++nc)
                acc[nc] = wmma_f16(a, bb[nc], acc[nc]);
            SCHED_FENCE();
        }
    }

#pragma unroll
    for (int nc = 0; nc < 8; ++nc)
#pragma unroll
        for (int v = 0; v < 8; ++v) {
            int row = m0 + v + 8 * lhalf;
            int col = nc * 16 + l16;
            float val = acc[nc][v] + b1s[col];
            h32[(size_t)row * EMB + col] = val;
            h16[(size_t)row * EMB + col] = (_Float16)val;
            h16t[(size_t)col * NR + row] = (_Float16)val;  // transposed copy
        }
}

// ---------------- d[i] = ||h_i||^2 ----------------
__global__ void __launch_bounds__(256)
k_rownorm(const float* __restrict__ h32, float* __restrict__ dv) {
    int row = blockIdx.x * 8 + (threadIdx.x >> 5);
    int lane = threadIdx.x & 31;
    v4f v = *(const v4f*)(h32 + (size_t)row * EMB + lane * 4);
    float s = v[0] * v[0] + v[1] * v[1] + v[2] * v[2] + v[3] * v[3];
#pragma unroll
    for (int off = 16; off; off >>= 1) s += __shfl_xor(s, off, 32);
    if (lane == 0) dv[row] = s;
}

// ---- fused agg = (2 h h^T / (d_i + d_j)) @ h, async double-buffered J-tiles ----
#define HJ_PITCH  136  // 272B rows: 16B aligned, bank-conflict-spreading
#define HJT_PITCH 40   // 80B rows
__global__ void __launch_bounds__(256)
k_agg(const _Float16* __restrict__ h16, const _Float16* __restrict__ h16t,
      const float* __restrict__ dv, float* __restrict__ agg) {
    __shared__ __align__(16) _Float16 hj[2][32 * HJ_PITCH];    // row-major J-tile
    __shared__ __align__(16) _Float16 hjt[2][128 * HJT_PITCH]; // transposed J-tile
    __shared__ __align__(16) _Float16 ptile[8][16 * 32];       // per-wave P staging
    __shared__ __align__(16) float djs[2][32];

    const int tid = threadIdx.x;
    const int lane = tid & 31, wave = tid >> 5;
    const int l16 = lane & 15, lhalf = lane >> 4;
    const int m0 = blockIdx.x * 128 + wave * 16;

    v16h aI[4];
#pragma unroll
    for (int kc = 0; kc < 4; ++kc)
        aI[kc] = load_a16(h16 + (size_t)m0 * EMB, EMB, l16, lhalf, kc * 32);
    float di[8];
#pragma unroll
    for (int v = 0; v < 8; ++v) di[v] = dv[m0 + v + 8 * lhalf];

    v8f acc[8];
#pragma unroll
    for (int nc = 0; nc < 8; ++nc) acc[nc] = zero8();

    // issue this thread's share of async copies for tile starting at row j
    auto stage = [&](int j, int buf) {
#pragma unroll
        for (int s = 0; s < 2; ++s) {
            int c = tid + s * 256;                 // 512 x 16B chunks each
            int row = c >> 4, off = (c & 15) * 8;  // hj: 32 rows x 16 chunks
            async_ld16(&hj[buf][row * HJ_PITCH + off],
                       h16 + (size_t)(j + row) * EMB + off);
            int col = c >> 2, off2 = (c & 3) * 8;  // hjt: 128 rows x 4 chunks
            async_ld16(&hjt[buf][col * HJT_PITCH + off2],
                       h16t + (size_t)col * NR + j + off2);
        }
        if (tid < 8) async_ld16(&djs[buf][tid * 4], dv + j + tid * 4);
    };

    stage(0, 0);
    wait_async();
    __syncthreads();

    for (int j = 0; j < NR; j += 32) {
        const int buf = (j >> 5) & 1;
        if (j + 32 < NR) stage(j + 32, buf ^ 1);   // overlap copy with compute

        const _Float16* hjb  = hj[buf];
        const _Float16* hjtb = hjt[buf];

        // ---- S tiles: batch all 16 B-frag loads, then 8 back-to-back WMMAs ----
        v16h bs[8];
#pragma unroll
        for (int kc = 0; kc < 4; ++kc) {
            bs[2 * kc]     = load_b(&hjb[(l16)      * HJ_PITCH + kc * 32 + 16 * lhalf]);
            bs[2 * kc + 1] = load_b(&hjb[(16 + l16) * HJ_PITCH + kc * 32 + 16 * lhalf]);
        }
        // issue the agg-phase B-frag loads now too: independent of S/P
        v16h bh[8];
#pragma unroll
        for (int nc = 0; nc < 8; ++nc)
            bh[nc] = load_b(&hjtb[(nc * 16 + l16) * HJT_PITCH + 16 * lhalf]);
        SCHED_FENCE();
        v8f s0 = zero8(), s1 = zero8();
#pragma unroll
        for (int kc = 0; kc < 4; ++kc) {
            s0 = wmma_f16(aI[kc], bs[2 * kc],     s0);
            s1 = wmma_f16(aI[kc], bs[2 * kc + 1], s1);
        }
        SCHED_FENCE();

        // ---- normalize: P = 2*S/(d_i + d_j), stage 16x32 f16 (per-wave) ----
        float dj0 = djs[buf][l16], dj1 = djs[buf][16 + l16];
        _Float16* prow = &ptile[wave][0];
#pragma unroll
        for (int v = 0; v < 8; ++v) {
            int m = v + 8 * lhalf;
            prow[m * 32 + l16]      = (_Float16)(2.0f * s0[v] / (di[v] + dj0));
            prow[m * 32 + 16 + l16] = (_Float16)(2.0f * s1[v] / (di[v] + dj1));
        }
        v16h aP = load_a16(prow, 32, l16, lhalf, 0);  // per-wave region: no barrier
        SCHED_FENCE();

        // ---- agg += P @ h_J: 8 back-to-back WMMAs ----
#pragma unroll
        for (int nc = 0; nc < 8; ++nc)
            acc[nc] = wmma_f16(aP, bh[nc], acc[nc]);
        SCHED_FENCE();

        wait_async();     // next tile's copies (issued above) have landed
        __syncthreads();  // everyone done reading buf before it is re-filled
    }

#pragma unroll
    for (int nc = 0; nc < 8; ++nc)
#pragma unroll
        for (int v = 0; v < 8; ++v)
            agg[(size_t)(m0 + v + 8 * lhalf) * EMB + nc * 16 + l16] = acc[nc][v];
}

// ---------------- out = relu(agg @ W2 + b2) ----------------
__global__ void __launch_bounds__(256)
k_out(const float* __restrict__ agg, const float* __restrict__ W2,
      const float* __restrict__ b2, float* __restrict__ out) {
    __shared__ __align__(16) _Float16 w2t[ODIM][EMB + 8];
    __shared__ float b2s[ODIM];

    const int tid = threadIdx.x;
    const int lane = tid & 31, wave = tid >> 5;
    const int l16 = lane & 15, lhalf = lane >> 4;
    const int m0 = blockIdx.x * 128 + wave * 16;

    if (tid < ODIM) b2s[tid] = b2[tid];
    for (int idx = tid; idx < ODIM * EMB; idx += 256) {
        int n = idx & 127;
        int k = idx >> 7;
        w2t[n][k] = (_Float16)W2[(size_t)k * ODIM + n];
    }
    __syncthreads();

    v8f acc[8];
#pragma unroll
    for (int nc = 0; nc < 8; ++nc) acc[nc] = zero8();
#pragma unroll
    for (int kc = 0; kc < 4; ++kc) {
        v16h a = load_a32(agg + (size_t)m0 * EMB, EMB, l16, lhalf, kc * 32);
        v16h bb[8];
#pragma unroll
        for (int nc = 0; nc < 8; ++nc)
            bb[nc] = load_b(&w2t[nc * 16 + l16][kc * 32 + 16 * lhalf]);
        SCHED_FENCE();
#pragma unroll
        for (int nc = 0; nc < 8; ++nc)
            acc[nc] = wmma_f16(a, bb[nc], acc[nc]);
        SCHED_FENCE();
    }

#pragma unroll
    for (int nc = 0; nc < 8; ++nc)
#pragma unroll
        for (int v = 0; v < 8; ++v) {
            int col = nc * 16 + l16;
            float val = acc[nc][v] + b2s[col];
            out[(size_t)(m0 + v + 8 * lhalf) * ODIM + col] = val > 0.0f ? val : 0.0f;
        }
}

extern "C" void kernel_launch(void* const* d_in, const int* in_sizes, int n_in,
                              void* d_out, int out_size, void* d_ws, size_t ws_size,
                              hipStream_t stream) {
    const float* x  = (const float*)d_in[0];
    const float* W1 = (const float*)d_in[1];
    const float* b1 = (const float*)d_in[2];
    const float* W2 = (const float*)d_in[3];
    const float* b2 = (const float*)d_in[4];

    char* ws = (char*)d_ws;
    float*    h32  = (float*)(ws);                                  // 8 MB
    _Float16* h16  = (_Float16*)(ws + ((size_t)8 << 20));           // 4 MB
    _Float16* h16t = (_Float16*)(ws + ((size_t)12 << 20));          // 4 MB
    float*    dv   = (float*)(ws + ((size_t)16 << 20));             // 64 KB
    float*    agg  = (float*)(ws + ((size_t)16 << 20) + 65536);     // 8 MB

    k_hidden <<<NR / 128, 256, 0, stream>>>(x, W1, b1, h32, h16, h16t);
    k_rownorm<<<NR / 8,   256, 0, stream>>>(h32, dv);
    k_agg    <<<NR / 128, 256, 0, stream>>>(h16, h16t, dv, agg);
    k_out    <<<NR / 128, 256, 0, stream>>>(agg, W2, b2, (float*)d_out);
}
